// SchEmb_15650860827293
// MI455X (gfx1250) — matrix operand order repaired and checked by reading, not compiled, not run on hardware
//
#include <hip/hip_runtime.h>
#include <hip/hip_bf16.h>

#define NN 10000
#define EE 320000
#define BBATCH 64
#define HH 128
#define FF 256
#define LLAYERS 5
#define NFFH 512
#define HP 80          // H - POS
#define PER_W (FF*HH + FF*FF + FF*HH + HH*FF + HH*HH)   // 180224 halfs per layer

typedef _Float16 v16h __attribute__((ext_vector_type(16)));
typedef _Float16 h8   __attribute__((ext_vector_type(8)));
typedef float    v8f  __attribute__((ext_vector_type(8)));
typedef float    f4v  __attribute__((ext_vector_type(4)));

// ---------------- WMMA helpers ----------------

static __device__ __forceinline__ v8f wmma16(v16h a, v16h b, v8f c) {
    // D = A(16x32,f16) * B(32x16,f16) + C(16x16,f32)
    return __builtin_amdgcn_wmma_f32_16x16x32_f16(false, a, false, b, (short)0, c, false, false);
}

// A-fragment (16x32 f16) from row-major [row][k] buffer (global or LDS).
// Lane l: row = row0 + (l&15); chunk0 k = k0 + 8*(l>>4); chunk1 k = chunk0 + 16.
static __device__ __forceinline__ v16h loadA_h(const _Float16* base, int ld,
                                               int row0, int k0, int lo, int hi) {
    const _Float16* p = base + (size_t)(row0 + lo) * ld + (k0 + 8 * hi);
    h8 c0 = *(const h8*)p;
    h8 c1 = *(const h8*)(p + 16);
    v16h a;
#pragma unroll
    for (int i = 0; i < 8; ++i) { a[i] = c0[i]; a[i + 8] = c1[i]; }
    return a;
}

// A-fragment with f32 -> f16 conversion on the fly (for agg matrix).
static __device__ __forceinline__ v16h loadA_f32(const float* base, int ld,
                                                 int row0, int k0, int lo, int hi) {
    const float* p = base + (size_t)(row0 + lo) * ld + (k0 + 8 * hi);
    f4v a0 = *(const f4v*)p;
    f4v a1 = *(const f4v*)(p + 4);
    f4v b0 = *(const f4v*)(p + 16);
    f4v b1 = *(const f4v*)(p + 20);
    v16h a;
#pragma unroll
    for (int i = 0; i < 4; ++i) {
        a[i]      = (_Float16)a0[i];
        a[i + 4]  = (_Float16)a1[i];
        a[i + 8]  = (_Float16)b0[i];
        a[i + 12] = (_Float16)b1[i];
    }
    return a;
}

// B-fragment (32x16 f16). Weights stored [n][k] row-major (ldk = K dimension).
// Lane l: n = n0 + (l&15); chunk0 k = k0 + 8*(l>>4); chunk1 k = chunk0 + 16.
static __device__ __forceinline__ v16h loadB_h(const _Float16* w, int ldk,
                                               int n0, int k0, int lo, int hi) {
    const _Float16* p = w + (size_t)(n0 + lo) * ldk + (k0 + 8 * hi);
    h8 c0 = *(const h8*)p;
    h8 c1 = *(const h8*)(p + 16);
    v16h b;
#pragma unroll
    for (int i = 0; i < 8; ++i) { b[i] = c0[i]; b[i + 8] = c1[i]; }
    return b;
}

static __device__ __forceinline__ float sspf(float v) {
    float sp = (v > 20.0f) ? v : log1pf(__expf(v));
    return sp - 0.69314718055994531f;   // shifted softplus
}

// ---------------- kernels ----------------

// Convert all per-layer weight matrices to f16, concatenated per layer:
// [w1(F,H) | w2(F,F) | lin1(F,H) | lin2(H,F) | lin(H,H)]
__global__ void k_wconv(const float* w1, const float* w2, const float* l1,
                        const float* l2, const float* lw, _Float16* out) {
    int idx = blockIdx.x * blockDim.x + threadIdx.x;
    if (idx >= LLAYERS * PER_W) return;
    int layer = idx / PER_W;
    int r = idx - layer * PER_W;
    float v;
    if (r < FF * HH)                       v = w1[(size_t)layer * FF * HH + r];
    else if (r < FF * HH + FF * FF)        v = w2[(size_t)layer * FF * FF + (r - FF * HH)];
    else if (r < 2 * FF * HH + FF * FF)    v = l1[(size_t)layer * FF * HH + (r - FF * HH - FF * FF)];
    else if (r < 3 * FF * HH + FF * FF)    v = l2[(size_t)layer * HH * FF + (r - 2 * FF * HH - FF * FF)];
    else                                   v = lw[(size_t)layer * HH * HH + (r - 3 * FF * HH - FF * FF)];
    out[idx] = (_Float16)v;
}

__global__ void k_zero(float* p, long long n) {
    long long i = (long long)blockIdx.x * blockDim.x + threadIdx.x;
    if (i < n) p[i] = 0.0f;
}

// h = concat(vert_emb[x], pos @ pos_w^T) ; also f16 copy for WMMA
__global__ void k_emb(const int* x, const float* vert, const float* pos,
                      const float* posw, float* h32, _Float16* h16) {
    int idx = blockIdx.x * blockDim.x + threadIdx.x;
    if (idx >= NN * HH) return;
    int n = idx >> 7, c = idx & 127;
    float v;
    if (c < HP) {
        v = vert[(size_t)x[n] * HP + c];
    } else {
        int j = c - HP;
        v = pos[n * 3 + 0] * posw[j * 3 + 0] +
            pos[n * 3 + 1] * posw[j * 3 + 1] +
            pos[n * 3 + 2] * posw[j * 3 + 2];
    }
    h32[idx] = v;
    h16[idx] = (_Float16)v;
}

// hf[N,F] = h16[N,H] @ lin1^T   (4 waves/block, 16-node tile, 64 cols per wave)
__global__ void k_node1(const _Float16* __restrict__ h16,
                        const _Float16* __restrict__ l1w,
                        float* __restrict__ hf) {
    int tile = blockIdx.x;                 // 625 tiles of 16 nodes
    int wave = threadIdx.x >> 5;
    int lane = threadIdx.x & 31, lo = lane & 15, hi = lane >> 4;
    int row0 = tile * 16;
    int nb0 = wave * 64;
    v8f acc[4] = {};
#pragma unroll
    for (int ks = 0; ks < 4; ++ks) {       // K = 128
        v16h a = loadA_h(h16, HH, row0, ks * 32, lo, hi);
#pragma unroll
        for (int nt = 0; nt < 4; ++nt)
            acc[nt] = wmma16(a, loadB_h(l1w, HH, nb0 + nt * 16, ks * 32, lo, hi), acc[nt]);
    }
#pragma unroll
    for (int nt = 0; nt < 4; ++nt) {
        int n = nb0 + nt * 16 + lo;
#pragma unroll
        for (int r = 0; r < 8; ++r) {
            int m = r + 8 * hi;
            hf[(size_t)(row0 + m) * FF + n] = acc[nt][r];
        }
    }
}

// Fused per-edge pipeline: ea tile -> ssp(ea@W1+b1) -> @W2+b2 -> * hf[src] -> atomic agg[dst]
__global__ void k_edge(const int* __restrict__ ei, const float* __restrict__ eattr,
                       const float* __restrict__ ew,
                       const _Float16* __restrict__ w1_16, const float* __restrict__ b1,
                       const _Float16* __restrict__ w2_16, const float* __restrict__ b2,
                       const float* __restrict__ hf, float* __restrict__ agg) {
    __shared__ __align__(32) _Float16 sEA[16 * 136];   // 16 edges x 128 (+8 pad)
    __shared__ __align__(32) _Float16 sT[16 * 264];    // 16 edges x 256 (+8 pad)
    __shared__ int sSrc[16], sDst[16];

    int tile = blockIdx.x;                 // 20000 tiles of 16 edges
    int e0 = tile * 16;
    int tid = threadIdx.x;

    if (tid < 16) { sSrc[tid] = ei[e0 + tid]; sDst[tid] = ei[EE + e0 + tid]; }
    // edge embedding tile: ea = edge_attr(E,4) @ edge_w(H,4)^T  (K=4 -> VALU)
    for (int t = tid; t < 16 * HH; t += 128) {
        int m = t >> 7, c = t & 127;
        const float* ea = eattr + (size_t)(e0 + m) * 4;
        const float* w = ew + c * 4;
        sEA[m * 136 + c] = (_Float16)(ea[0]*w[0] + ea[1]*w[1] + ea[2]*w[2] + ea[3]*w[3]);
    }
    __syncthreads();

    int wave = tid >> 5, lane = tid & 31, lo = lane & 15, hi = lane >> 4;
    int nb0 = wave * 64;

    {   // stage 1: t = ssp(ea @ w1^T + b1) -> LDS f16
        v8f acc[4] = {};
#pragma unroll
        for (int ks = 0; ks < 4; ++ks) {   // K = 128
            v16h a = loadA_h(sEA, 136, 0, ks * 32, lo, hi);
#pragma unroll
            for (int nt = 0; nt < 4; ++nt)
                acc[nt] = wmma16(a, loadB_h(w1_16, HH, nb0 + nt * 16, ks * 32, lo, hi), acc[nt]);
        }
#pragma unroll
        for (int nt = 0; nt < 4; ++nt) {
            int n = nb0 + nt * 16 + lo;
            float bias = b1[n];
#pragma unroll
            for (int r = 0; r < 8; ++r) {
                int m = r + 8 * hi;
                sT[m * 264 + n] = (_Float16)sspf(acc[nt][r] + bias);
            }
        }
    }
    __syncthreads();

    {   // stage 2: W = t @ w2^T + b2, then modulate+scatter
        v8f acc[4] = {};
#pragma unroll
        for (int ks = 0; ks < 8; ++ks) {   // K = 256
            v16h a = loadA_h(sT, 264, 0, ks * 32, lo, hi);
#pragma unroll
            for (int nt = 0; nt < 4; ++nt)
                acc[nt] = wmma16(a, loadB_h(w2_16, FF, nb0 + nt * 16, ks * 32, lo, hi), acc[nt]);
        }
#pragma unroll
        for (int nt = 0; nt < 4; ++nt) {
            int n = nb0 + nt * 16 + lo;
            float bias = b2[n];
#pragma unroll
            for (int r = 0; r < 8; ++r) {
                int m = r + 8 * hi;
                float Wv = acc[nt][r] + bias;
                float msg = Wv * hf[(size_t)sSrc[m] * FF + n];
                atomicAdd(&agg[(size_t)sDst[m] * FF + n], msg);
            }
        }
    }
}

// out = ssp(agg@lin2^T+b) @ lin^T + b ; h = relu(out)+h  (2 waves/block, 16-node tile)
__global__ void k_node2(const float* __restrict__ agg,
                        const _Float16* __restrict__ l2_16, const float* __restrict__ l2b,
                        const _Float16* __restrict__ lw_16, const float* __restrict__ lb,
                        float* __restrict__ h32, _Float16* __restrict__ h16) {
    __shared__ __align__(32) _Float16 sT[16 * 136];    // 16 nodes x 128 (+8 pad)
    int tile = blockIdx.x;
    int row0 = tile * 16;
    int tid = threadIdx.x;
    int wave = tid >> 5, lane = tid & 31, lo = lane & 15, hi = lane >> 4;
    int nb0 = wave * 64;

    {   // stage 1: K = 256 over agg (f32 -> f16 convert in regs)
        v8f acc[4] = {};
#pragma unroll
        for (int ks = 0; ks < 8; ++ks) {
            v16h a = loadA_f32(agg, FF, row0, ks * 32, lo, hi);
#pragma unroll
            for (int nt = 0; nt < 4; ++nt)
                acc[nt] = wmma16(a, loadB_h(l2_16, FF, nb0 + nt * 16, ks * 32, lo, hi), acc[nt]);
        }
#pragma unroll
        for (int nt = 0; nt < 4; ++nt) {
            int n = nb0 + nt * 16 + lo;
            float bias = l2b[n];
#pragma unroll
            for (int r = 0; r < 8; ++r) {
                int m = r + 8 * hi;
                sT[m * 136 + n] = (_Float16)sspf(acc[nt][r] + bias);
            }
        }
    }
    __syncthreads();

    {   // stage 2: K = 128 ; relu + residual update of h32/h16
        v8f acc[4] = {};
#pragma unroll
        for (int ks = 0; ks < 4; ++ks) {
            v16h a = loadA_h(sT, 136, 0, ks * 32, lo, hi);
#pragma unroll
            for (int nt = 0; nt < 4; ++nt)
                acc[nt] = wmma16(a, loadB_h(lw_16, HH, nb0 + nt * 16, ks * 32, lo, hi), acc[nt]);
        }
#pragma unroll
        for (int nt = 0; nt < 4; ++nt) {
            int n = nb0 + nt * 16 + lo;
            float bias = lb[n];
#pragma unroll
            for (int r = 0; r < 8; ++r) {
                int m = r + 8 * hi;
                float o = acc[nt][r] + bias;
                o = o > 0.0f ? o : 0.0f;
                size_t idx = (size_t)(row0 + m) * HH + n;
                float hn = o + h32[idx];
                h32[idx] = hn;
                h16[idx] = (_Float16)hn;
            }
        }
    }
}

// global mean-pool accumulation (atomics into pooled/cnt)
__global__ void k_pool(const float* __restrict__ h32, const int* __restrict__ batch,
                       float* __restrict__ pooled, float* __restrict__ cnt) {
    int idx = blockIdx.x * blockDim.x + threadIdx.x;
    if (idx >= NN * HH) return;
    int n = idx >> 7, c = idx & 127;
    int b = batch[n];
    atomicAdd(&pooled[b * HH + c], h32[idx]);
    if (c == 0) atomicAdd(&cnt[b], 1.0f);
}

// head: gelu(pooled/cnt @ w1^T + b1) @ w2^T + b2  (tiny -> VALU)
__global__ void k_head(const float* __restrict__ pooled, const float* __restrict__ cnt,
                       const float* __restrict__ w1, const float* __restrict__ b1,
                       const float* __restrict__ w2, const float* __restrict__ b2,
                       float* __restrict__ out) {
    __shared__ float sMean[HH];
    __shared__ float sRed[128];
    int b = blockIdx.x, t = threadIdx.x;
    float c = fmaxf(cnt[b], 1.0f);
    sMean[t] = pooled[b * HH + t] / c;
    __syncthreads();
    float partial = 0.0f;
#pragma unroll
    for (int q = 0; q < 4; ++q) {
        int j = t + 128 * q;                    // NFF = 512
        float s = b1[j];
        for (int k = 0; k < HH; ++k) s += sMean[k] * w1[j * HH + k];
        float g = 0.5f * s * (1.0f + erff(s * 0.70710678118654752f));
        partial += g * w2[j];
    }
    sRed[t] = partial;
    __syncthreads();
    for (int st = 64; st > 0; st >>= 1) {
        if (t < st) sRed[t] += sRed[t + st];
        __syncthreads();
    }
    if (t == 0) out[b] = sRed[0] + b2[0];
}

// ---------------- host ----------------

extern "C" void kernel_launch(void* const* d_in, const int* in_sizes, int n_in,
                              void* d_out, int out_size, void* d_ws, size_t ws_size,
                              hipStream_t stream) {
    const int*   x      = (const int*)d_in[0];
    const int*   ei     = (const int*)d_in[1];
    const float* eattr  = (const float*)d_in[2];
    const int*   batch  = (const int*)d_in[3];
    const float* pos    = (const float*)d_in[4];
    const float* vert   = (const float*)d_in[5];
    const float* posw   = (const float*)d_in[6];
    const float* ew     = (const float*)d_in[7];
    const float* mlp_w1 = (const float*)d_in[8];
    const float* mlp_b1 = (const float*)d_in[9];
    const float* mlp_w2 = (const float*)d_in[10];
    const float* mlp_b2 = (const float*)d_in[11];
    const float* lin1_w = (const float*)d_in[12];
    const float* lin2_w = (const float*)d_in[13];
    const float* lin2_b = (const float*)d_in[14];
    const float* lin_w  = (const float*)d_in[15];
    const float* lin_b  = (const float*)d_in[16];
    const float* hw1    = (const float*)d_in[17];
    const float* hb1    = (const float*)d_in[18];
    const float* hw2    = (const float*)d_in[19];
    const float* hb2    = (const float*)d_in[20];
    float* out = (float*)d_out;

    char* ws = (char*)d_ws;
    size_t off = 0;
    auto alloc = [&](size_t bytes) -> void* {
        void* p = ws + off;
        off = (off + bytes + 255) & ~(size_t)255;
        return p;
    };
    float*     h32    = (float*)alloc((size_t)NN * HH * 4);
    float*     hf     = (float*)alloc((size_t)NN * FF * 4);
    float*     agg    = (float*)alloc((size_t)NN * FF * 4);
    float*     pooled = (float*)alloc((size_t)BBATCH * HH * 4);   // 32768B, 256-aligned
    float*     cnt    = (float*)alloc((size_t)BBATCH * 4);        // contiguous after pooled
    _Float16*  h16    = (_Float16*)alloc((size_t)NN * HH * 2);
    _Float16*  w16    = (_Float16*)alloc((size_t)LLAYERS * PER_W * 2);

    k_wconv<<<(LLAYERS * PER_W + 255) / 256, 256, 0, stream>>>(mlp_w1, mlp_w2, lin1_w,
                                                               lin2_w, lin_w, w16);
    k_emb<<<(NN * HH + 255) / 256, 256, 0, stream>>>(x, vert, pos, posw, h32, h16);

    for (int i = 0; i < LLAYERS; ++i) {
        const _Float16* lw16  = w16 + (size_t)i * PER_W;
        const _Float16* w1_16 = lw16;
        const _Float16* w2_16 = w1_16 + FF * HH;
        const _Float16* l1_16 = w2_16 + FF * FF;
        const _Float16* l2_16 = l1_16 + FF * HH;
        const _Float16* lm_16 = l2_16 + HH * FF;

        k_node1<<<NN / 16, 128, 0, stream>>>(h16, l1_16, hf);
        k_zero<<<((long long)NN * FF + 255) / 256, 256, 0, stream>>>(agg, (long long)NN * FF);
        k_edge<<<EE / 16, 128, 0, stream>>>(ei, eattr, ew, w1_16, mlp_b1 + i * FF,
                                            w2_16, mlp_b2 + i * FF, hf, agg);
        k_node2<<<NN / 16, 64, 0, stream>>>(agg, l2_16, lin2_b + i * HH,
                                            lm_16, lin_b + i * HH, h32, h16);
    }

    k_zero<<<(BBATCH * HH + BBATCH + 255) / 256, 256, 0, stream>>>(pooled,
                                                                   BBATCH * HH + BBATCH);
    k_pool<<<(NN * HH + 255) / 256, 256, 0, stream>>>(h32, batch, pooled, cnt);
    k_head<<<BBATCH, 128, 0, stream>>>(pooled, cnt, hw1, hb1, hw2, hb2, out);

    (void)in_sizes; (void)n_in; (void)out_size; (void)ws_size;
}
